// ExpertChoiceFFN_74354473828466
// MI455X (gfx1250) — matrix operand order repaired and checked by reading, not compiled, not run on hardware
//
#include <hip/hip_runtime.h>
#include <hip/hip_bf16.h>
#include <math.h>

typedef _Float16 half8 __attribute__((ext_vector_type(8)));
typedef _Float16 v16h  __attribute__((ext_vector_type(16)));
typedef float    v8f   __attribute__((ext_vector_type(8)));

#define NTOK 16384   // B*T
#define DM   1024
#define DFF  4096
#define NE   8
#define CAP  2560    // ceil(1.25*NTOK/NE)
#define KPAD 40      // LDS row stride in halfs (80B) -> conflict-free b128 reads

static __device__ __forceinline__ float atomic_add_f32(float* p, float v) {
  return __hip_atomic_fetch_add(p, v, __ATOMIC_RELAXED, __HIP_MEMORY_SCOPE_AGENT);
}

// CDNA5 async global->LDS copy (ASYNCcnt path). Generic LDS pointer's low 32
// bits are the LDS byte offset (ISA 10.2 aperture rule).
static __device__ __forceinline__ void async_copy_b128(const _Float16* gsrc,
                                                       _Float16* ldst) {
  unsigned int l = (unsigned int)(uintptr_t)ldst;
  asm volatile("global_load_async_to_lds_b128 %0, %1, off"
               :: "v"(l), "v"(gsrc) : "memory");
}
static __device__ __forceinline__ void async_wait0() {
  asm volatile("s_wait_asynccnt 0" ::: "memory");
}

// ---------------------------------------------------------------- zero output
__global__ void zero_f32(float* p, size_t n) {
  size_t i = (size_t)blockIdx.x * blockDim.x + threadIdx.x;
  size_t stride = (size_t)gridDim.x * blockDim.x;
  for (; i < n; i += stride) p[i] = 0.0f;
}

// ------------------------------------------------- router: logits, softmax,
// probs (N,E) to ws, aux-loss (mean sum p*log(p+eps)) atomically to *aux.
__global__ void __launch_bounds__(128)
router_kernel(const float* __restrict__ hidden, const float* __restrict__ rw,
              float* __restrict__ probs, float* __restrict__ aux) {
  __shared__ float wLds[DM * NE];     // 32 KB
  __shared__ float hLds[4][DM];       // 16 KB
  __shared__ float partial[4][32];
  __shared__ float lg[4][NE];
  const int tid = threadIdx.x;
  const int tok0 = blockIdx.x * 4;
  for (int i = tid; i < DM * NE; i += 128) wLds[i] = rw[i];
  for (int i = tid; i < 4 * DM; i += 128) {
    int tl = i >> 10, d = i & (DM - 1);
    hLds[tl][d] = hidden[(size_t)(tok0 + tl) * DM + d];
  }
  __syncthreads();
  const int wave = tid >> 5, lane = tid & 31;
  const int e = lane & 7, chunk = lane >> 3;
  float acc = 0.0f;
  for (int d = chunk * 256; d < chunk * 256 + 256; ++d)
    acc += hLds[wave][d] * wLds[d * NE + e];
  partial[wave][lane] = acc;
  __syncthreads();
  if (tid < 4 * NE) {
    int tl = tid >> 3, ee = tid & 7;
    lg[tl][ee] = partial[tl][ee] + partial[tl][8 + ee] +
                 partial[tl][16 + ee] + partial[tl][24 + ee];
  }
  __syncthreads();
  if (tid < 4) {
    int tl = tid;
    float m = lg[tl][0];
    for (int j = 1; j < NE; ++j) m = fmaxf(m, lg[tl][j]);
    float ex[NE], sum = 0.0f;
    for (int j = 0; j < NE; ++j) { ex[j] = expf(lg[tl][j] - m); sum += ex[j]; }
    float inv = 1.0f / sum, s = 0.0f;
    for (int j = 0; j < NE; ++j) {
      float p = ex[j] * inv;
      probs[(size_t)(tok0 + tl) * NE + j] = p;
      s += p * logf(p + 1e-8f);
    }
    atomic_add_f32(aux, s * (1.0f / (float)NTOK));
  }
}

// ------------------------------------------------- exact top-C per expert via
// 3-round radix select on float bit patterns (probs > 0 so bits are ordered).
__global__ void __launch_bounds__(1024)
topk_kernel(const float* __restrict__ probs,
            int* __restrict__ sel_idx, float* __restrict__ sel_w) {
  __shared__ unsigned int hist[2048];
  __shared__ unsigned int sbuf[2][2048];
  __shared__ unsigned int sh_b, sh_above;
  __shared__ int cnt_gt, cnt_eq;
  const int e = blockIdx.x;
  const int tid = threadIdx.x;
  const int nt = blockDim.x;  // 1024
  unsigned int pref = 0u, prefMask = 0u, R = CAP;
  const int shifts[3] = {21, 10, 0};
  const int dbits[3]  = {11, 11, 10};
  for (int r = 0; r < 3; ++r) {
    const int shift = shifts[r];
    const unsigned int nb = 1u << dbits[r];
    const unsigned int dmask = nb - 1u;
    for (unsigned int b = tid; b < nb; b += nt) hist[b] = 0u;
    __syncthreads();
    for (int i = tid; i < NTOK; i += nt) {
      unsigned int k = __float_as_uint(probs[(size_t)i * NE + e]);
      if ((k & prefMask) == pref) atomicAdd(&hist[(k >> shift) & dmask], 1u);
    }
    __syncthreads();
    int cur = 0;
    for (unsigned int b = tid; b < nb; b += nt) sbuf[0][b] = hist[b];
    __syncthreads();
    for (unsigned int off = 1; off < nb; off <<= 1) {
      for (unsigned int b = tid; b < nb; b += nt) {
        unsigned int v = sbuf[cur][b];
        if (b + off < nb) v += sbuf[cur][b + off];
        sbuf[1 - cur][b] = v;
      }
      cur = 1 - cur;
      __syncthreads();
    }
    for (unsigned int b = tid; b < nb; b += nt) {
      unsigned int s = sbuf[cur][b];
      unsigned int sn = (b + 1 < nb) ? sbuf[cur][b + 1] : 0u;
      if (s >= R && sn < R) { sh_b = b; sh_above = sn; }
    }
    __syncthreads();
    pref |= (sh_b << shift);
    prefMask |= (dmask << (unsigned)shift);
    R -= sh_above;
    __syncthreads();
  }
  const unsigned int T = pref;
  if (tid == 0) { cnt_gt = 0; cnt_eq = 0; }
  __syncthreads();
  const unsigned int G = CAP - R;  // count of keys strictly > T
  for (int i = tid; i < NTOK; i += nt) {
    unsigned int k = __float_as_uint(probs[(size_t)i * NE + e]);
    if (k > T) {
      int p = atomicAdd(&cnt_gt, 1);
      sel_idx[e * CAP + p] = i;
      sel_w[e * CAP + p] = __uint_as_float(k);
    } else if (k == T) {
      int q = atomicAdd(&cnt_eq, 1);
      if ((unsigned int)q < R) {
        int p = (int)G + q;
        sel_idx[e * CAP + p] = i;
        sel_w[e * CAP + p] = __uint_as_float(k);
      }
    }
  }
}

// ------------------------------------------------- weight pre-pass:
// dst[e][c][r] = (f16) src[e][r][c]  (B matrices become [n][k]-major).
__global__ void __launch_bounds__(256)
transpose_cvt_kernel(const float* __restrict__ src, _Float16* __restrict__ dst,
                     int R, int Ccols) {
  __shared__ float t[32][33];
  const size_t eo = (size_t)blockIdx.z * (size_t)R * Ccols;
  const float* s = src + eo;
  _Float16* d = dst + eo;
  const int cb = blockIdx.x * 32, rb = blockIdx.y * 32;
  const int tx = threadIdx.x, ty = threadIdx.y;
  for (int j = 0; j < 4; ++j)
    t[ty + j * 8][tx] = s[(size_t)(rb + ty + j * 8) * Ccols + cb + tx];
  __syncthreads();
  for (int j = 0; j < 4; ++j)
    d[(size_t)(cb + ty + j * 8) * R + rb + tx] = (_Float16)t[tx][ty + j * 8];
}

static __device__ __forceinline__ v16h frag16(half8 lo, half8 hi) {
  return __builtin_shufflevector(lo, hi, 0, 1, 2, 3, 4, 5, 6, 7,
                                 8, 9, 10, 11, 12, 13, 14, 15);
}

// ------------------------------------------------- GEMM1: H = gelu(gather(X) @ W1)
// 128x64 block tile, BK=32, double-buffered LDS (1 barrier/iter), 8 waves,
// 32x32 per wave => 4 WMMA per 8 DS fragment loads. B staged via async->LDS.
__global__ void __launch_bounds__(256)
gemm1_kernel(const float* __restrict__ hidden, const _Float16* __restrict__ W1t,
             const int* __restrict__ sel_idx, _Float16* __restrict__ H16) {
  __shared__ __align__(16) _Float16 As[2 * 128 * KPAD];  // 20 KB
  __shared__ __align__(16) _Float16 Bs[2 * 64 * KPAD];   // 10 KB
  __shared__ int toks[128];
  const int e = blockIdx.z, m0 = blockIdx.y * 128, n0 = blockIdx.x * 64;
  const int tid = threadIdx.x;
  if (tid < 128) toks[tid] = sel_idx[e * CAP + m0 + tid];
  const int ar = tid >> 1, akq = (tid & 1) * 16;   // A staging: 16 f32 / thread
  const int br = tid >> 2, bkq = (tid & 3) * 8;    // B staging: 1 b128 / thread
  const int wave = tid >> 5, lane = tid & 31;
  const int wm = wave >> 1, wn = wave & 1;
  const int arow = wm * 32 + (lane & 15);
  const int koffA = (lane >> 4) * 8;     // ISA A layout: hi half-wave K+8
  const int bcol = wn * 32 + (lane & 15);
  const int koffB = (lane >> 4) * 8;     // ISA B layout: hi half-wave K+8
  const _Float16* W1e = W1t + (size_t)e * DFF * DM;
  v8f acc[2][2] = {};
  __syncthreads();

  // prologue: stage kb=0 into buffer 0
  {
    const float* src = hidden + (size_t)toks[ar] * DM + akq;
#pragma unroll
    for (int q = 0; q < 2; ++q) {
      float4 f0 = *(const float4*)(src + q * 8);
      float4 f1 = *(const float4*)(src + q * 8 + 4);
      half8 h;
      h[0] = (_Float16)f0.x; h[1] = (_Float16)f0.y;
      h[2] = (_Float16)f0.z; h[3] = (_Float16)f0.w;
      h[4] = (_Float16)f1.x; h[5] = (_Float16)f1.y;
      h[6] = (_Float16)f1.z; h[7] = (_Float16)f1.w;
      *(half8*)&As[ar * KPAD + akq + q * 8] = h;
    }
    async_copy_b128(W1e + (size_t)(n0 + br) * DM + bkq, &Bs[br * KPAD + bkq]);
    async_wait0();
  }
  __syncthreads();

  int p = 0;
  for (int kb = 0; kb < DM; kb += 32) {
    const _Float16* Ap = As + p * (128 * KPAD);
    const _Float16* Bp = Bs + p * (64 * KPAD);
    v16h aF0 = frag16(*(const half8*)&Ap[arow * KPAD + koffA],
                      *(const half8*)&Ap[arow * KPAD + 16 + koffA]);
    v16h aF1 = frag16(*(const half8*)&Ap[(arow + 16) * KPAD + koffA],
                      *(const half8*)&Ap[(arow + 16) * KPAD + 16 + koffA]);
    v16h bF0 = frag16(*(const half8*)&Bp[bcol * KPAD + koffB],
                      *(const half8*)&Bp[bcol * KPAD + koffB + 16]);
    v16h bF1 = frag16(*(const half8*)&Bp[(bcol + 16) * KPAD + koffB],
                      *(const half8*)&Bp[(bcol + 16) * KPAD + koffB + 16]);
    const bool more = (kb + 32) < DM;
    if (more) {  // stage next K block into the other buffer
      _Float16* An = As + (p ^ 1) * (128 * KPAD);
      _Float16* Bn = Bs + (p ^ 1) * (64 * KPAD);
      const float* src = hidden + (size_t)toks[ar] * DM + kb + 32 + akq;
#pragma unroll
      for (int q = 0; q < 2; ++q) {
        float4 f0 = *(const float4*)(src + q * 8);
        float4 f1 = *(const float4*)(src + q * 8 + 4);
        half8 h;
        h[0] = (_Float16)f0.x; h[1] = (_Float16)f0.y;
        h[2] = (_Float16)f0.z; h[3] = (_Float16)f0.w;
        h[4] = (_Float16)f1.x; h[5] = (_Float16)f1.y;
        h[6] = (_Float16)f1.z; h[7] = (_Float16)f1.w;
        *(half8*)&An[ar * KPAD + akq + q * 8] = h;
      }
      async_copy_b128(W1e + (size_t)(n0 + br) * DM + kb + 32 + bkq,
                      &Bn[br * KPAD + bkq]);
    }
    acc[0][0] = __builtin_amdgcn_wmma_f32_16x16x32_f16(false, aF0, false, bF0,
                                                       (short)0, acc[0][0], false, false);
    acc[0][1] = __builtin_amdgcn_wmma_f32_16x16x32_f16(false, aF0, false, bF1,
                                                       (short)0, acc[0][1], false, false);
    acc[1][0] = __builtin_amdgcn_wmma_f32_16x16x32_f16(false, aF1, false, bF0,
                                                       (short)0, acc[1][0], false, false);
    acc[1][1] = __builtin_amdgcn_wmma_f32_16x16x32_f16(false, aF1, false, bF1,
                                                       (short)0, acc[1][1], false, false);
    if (more) async_wait0();
    __syncthreads();
    p ^= 1;
  }

  const int cBase = n0 + wn * 32 + (lane & 15);
  const int rbase = (lane >> 4) * 8;
  _Float16* He = H16 + (size_t)e * CAP * DFF;
  for (int tm = 0; tm < 2; ++tm)
    for (int tn = 0; tn < 2; ++tn) {
      const int mB = m0 + wm * 32 + tm * 16 + rbase;
      const int c = cBase + tn * 16;
#pragma unroll
      for (int i = 0; i < 8; ++i) {
        float x = acc[tm][tn][i];
        float g = 0.5f * x * (1.0f + erff(x * 0.70710678118654752f));
        He[(size_t)(mB + i) * DFF + c] = (_Float16)g;
      }
    }
}

// ------------------------------------------------- GEMM2: out += w*(H @ W2), scatter.
// Both tiles are f16 copies -> fully async global->LDS staging.
__global__ void __launch_bounds__(256)
gemm2_kernel(const _Float16* __restrict__ H16, const _Float16* __restrict__ W2t,
             const int* __restrict__ sel_idx, const float* __restrict__ sel_w,
             float* __restrict__ out) {
  __shared__ __align__(16) _Float16 As[2 * 128 * KPAD];
  __shared__ __align__(16) _Float16 Bs[2 * 64 * KPAD];
  __shared__ int toks[128];
  __shared__ float sw[128];
  const int e = blockIdx.z, m0 = blockIdx.y * 128, n0 = blockIdx.x * 64;
  const int tid = threadIdx.x;
  if (tid < 128) {
    toks[tid] = sel_idx[e * CAP + m0 + tid];
    sw[tid] = sel_w[e * CAP + m0 + tid];
  }
  const int ar = tid >> 1, akq = (tid & 1) * 16;   // A staging: 2 b128 / thread
  const int br = tid >> 2, bkq = (tid & 3) * 8;    // B staging: 1 b128 / thread
  const int wave = tid >> 5, lane = tid & 31;
  const int wm = wave >> 1, wn = wave & 1;
  const int arow = wm * 32 + (lane & 15);
  const int koffA = (lane >> 4) * 8;
  const int bcol = wn * 32 + (lane & 15);
  const int koffB = (lane >> 4) * 8;
  const _Float16* He = H16 + (size_t)e * CAP * DFF;
  const _Float16* W2e = W2t + (size_t)e * DM * DFF;
  v8f acc[2][2] = {};

  {   // prologue: stage kb=0 into buffer 0 (all async)
    async_copy_b128(He + (size_t)(m0 + ar) * DFF + akq, &As[ar * KPAD + akq]);
    async_copy_b128(He + (size_t)(m0 + ar) * DFF + akq + 8,
                    &As[ar * KPAD + akq + 8]);
    async_copy_b128(W2e + (size_t)(n0 + br) * DFF + bkq, &Bs[br * KPAD + bkq]);
    async_wait0();
  }
  __syncthreads();

  int p = 0;
  for (int kb = 0; kb < DFF; kb += 32) {
    const _Float16* Ap = As + p * (128 * KPAD);
    const _Float16* Bp = Bs + p * (64 * KPAD);
    v16h aF0 = frag16(*(const half8*)&Ap[arow * KPAD + koffA],
                      *(const half8*)&Ap[arow * KPAD + 16 + koffA]);
    v16h aF1 = frag16(*(const half8*)&Ap[(arow + 16) * KPAD + koffA],
                      *(const half8*)&Ap[(arow + 16) * KPAD + 16 + koffA]);
    v16h bF0 = frag16(*(const half8*)&Bp[bcol * KPAD + koffB],
                      *(const half8*)&Bp[bcol * KPAD + koffB + 16]);
    v16h bF1 = frag16(*(const half8*)&Bp[(bcol + 16) * KPAD + koffB],
                      *(const half8*)&Bp[(bcol + 16) * KPAD + koffB + 16]);
    const bool more = (kb + 32) < DFF;
    if (more) {
      _Float16* An = As + (p ^ 1) * (128 * KPAD);
      _Float16* Bn = Bs + (p ^ 1) * (64 * KPAD);
      const _Float16* as = He + (size_t)(m0 + ar) * DFF + kb + 32 + akq;
      async_copy_b128(as, &An[ar * KPAD + akq]);
      async_copy_b128(as + 8, &An[ar * KPAD + akq + 8]);
      async_copy_b128(W2e + (size_t)(n0 + br) * DFF + kb + 32 + bkq,
                      &Bn[br * KPAD + bkq]);
    }
    acc[0][0] = __builtin_amdgcn_wmma_f32_16x16x32_f16(false, aF0, false, bF0,
                                                       (short)0, acc[0][0], false, false);
    acc[0][1] = __builtin_amdgcn_wmma_f32_16x16x32_f16(false, aF0, false, bF1,
                                                       (short)0, acc[0][1], false, false);
    acc[1][0] = __builtin_amdgcn_wmma_f32_16x16x32_f16(false, aF1, false, bF0,
                                                       (short)0, acc[1][0], false, false);
    acc[1][1] = __builtin_amdgcn_wmma_f32_16x16x32_f16(false, aF1, false, bF1,
                                                       (short)0, acc[1][1], false, false);
    if (more) async_wait0();
    __syncthreads();
    p ^= 1;
  }

  const int cBase = n0 + wn * 32 + (lane & 15);
  const int rbase = (lane >> 4) * 8;
  for (int tm = 0; tm < 2; ++tm)
    for (int tn = 0; tn < 2; ++tn) {
      const int l0 = wm * 32 + tm * 16 + rbase;
      const int c = cBase + tn * 16;
#pragma unroll
      for (int i = 0; i < 8; ++i) {
        const int lrow = l0 + i;
        atomic_add_f32(&out[(size_t)toks[lrow] * DM + c],
                       acc[tm][tn][i] * sw[lrow]);
      }
    }
}

// ---------------------------------------------------------------- launcher
extern "C" void kernel_launch(void* const* d_in, const int* in_sizes, int n_in,
                              void* d_out, int out_size, void* d_ws, size_t ws_size,
                              hipStream_t stream) {
  (void)in_sizes; (void)n_in; (void)out_size; (void)ws_size;
  const float* hidden = (const float*)d_in[0];   // (B,T,D)
  const float* rw     = (const float*)d_in[1];   // (D,E)
  const float* W1     = (const float*)d_in[2];   // (E,D,F)
  const float* W2     = (const float*)d_in[3];   // (E,F,D)
  float* out = (float*)d_out;                    // (B,T,D) flat + aux scalar

  char* ws = (char*)d_ws;
  float*    probs   = (float*)(ws);                              // 512 KB
  int*      sel_idx = (int*)(ws + 524288);                       // 80 KB
  float*    sel_w   = (float*)(ws + 606208);                     // 80 KB
  _Float16* W1t     = (_Float16*)(ws + 688128);                  // 64 MB (E,F,D) f16
  _Float16* W2t     = (_Float16*)(ws + 688128 + 67108864ull);    // 64 MB (E,D,F) f16
  _Float16* H16     = (_Float16*)(ws + 688128 + 134217728ull);   // 160 MB (E,C,F) f16

  const size_t outElems = (size_t)NTOK * DM + 1;  // output + aux slot
  zero_f32<<<2048, 256, 0, stream>>>(out, outElems);
  router_kernel<<<NTOK / 4, 128, 0, stream>>>(hidden, rw, probs,
                                              out + (size_t)NTOK * DM);
  topk_kernel<<<NE, 1024, 0, stream>>>(probs, sel_idx, sel_w);
  transpose_cvt_kernel<<<dim3(DFF / 32, DM / 32, NE), dim3(32, 8), 0, stream>>>(
      W1, W1t, DM, DFF);
  transpose_cvt_kernel<<<dim3(DM / 32, DFF / 32, NE), dim3(32, 8), 0, stream>>>(
      W2, W2t, DFF, DM);
  gemm1_kernel<<<dim3(DFF / 64, CAP / 128, NE), 256, 0, stream>>>(hidden, W1t,
                                                                  sel_idx, H16);
  gemm2_kernel<<<dim3(DM / 64, CAP / 128, NE), 256, 0, stream>>>(H16, W2t,
                                                                 sel_idx, sel_w, out);
}
